// AttentionHead_50354196578492
// MI455X (gfx1250) — matrix operand order repaired and checked by reading, not compiled, not run on hardware
//
#include <hip/hip_runtime.h>

// B=8, L=2048, D=1024, H=64
#define B_ 8
#define L_ 2048
#define D_ 1024
#define H_ 64

typedef _Float16 h4   __attribute__((ext_vector_type(4)));
typedef _Float16 h8   __attribute__((ext_vector_type(8)));
typedef _Float16 v16h __attribute__((ext_vector_type(16)));
typedef float    v8f  __attribute__((ext_vector_type(8)));

union AF { v16h v; h8 h[2]; };

static __device__ __forceinline__ v8f zero8() {
    v8f z;
    #pragma unroll
    for (int i = 0; i < 8; ++i) z[i] = 0.0f;
    return z;
}

static __device__ __forceinline__ v8f wmma_f16(v16h a, v16h b, v8f c) {
    // (neg_a, A, neg_b, B, c_mod, C, reuse_a, reuse_b)
    return __builtin_amdgcn_wmma_f32_16x16x32_f16(false, a, false, b, (short)0, c, false, false);
}

// ---- async global->LDS copy of 16 bytes per lane (CDNA5 ASYNCcnt path) ----
#if defined(__gfx1250__) && __has_builtin(__builtin_amdgcn_global_load_async_to_lds_b128)
typedef int v4i_vs __attribute__((vector_size(16)));   // GCC-style int4, per builtin proto
static __device__ __forceinline__ void async_cp16(void* lds, const void* g) {
    __builtin_amdgcn_global_load_async_to_lds_b128(
        (__attribute__((address_space(1))) v4i_vs*)g,
        (__attribute__((address_space(3))) v4i_vs*)lds, 0, 0);
}
#else
static __device__ __forceinline__ void async_cp16(void* lds, const void* g) {
    *(h8*)lds = *(const h8*)g;      // synchronous fallback
}
#endif

static __device__ __forceinline__ void wait_async0() {
#if defined(__gfx1250__)
#if __has_builtin(__builtin_amdgcn_s_wait_asynccnt)
    __builtin_amdgcn_s_wait_asynccnt(0);
#else
    asm volatile("s_wait_asynccnt 0x0" ::: "memory");
#endif
#endif
}

// ---------------------------------------------------------------------------
// Kernel 0: one-time weight conversion: W[k][h] f32 -> WT[w][h][k] f16.
// 196608 elements; grid 768 x 256.
// ---------------------------------------------------------------------------
__global__ __launch_bounds__(256) void convert_w_kernel(
    const float* __restrict__ Wq, const float* __restrict__ Wk,
    const float* __restrict__ Wv, _Float16* __restrict__ wt)
{
    int gid = blockIdx.x * 256 + threadIdx.x;   // (w*64 + h)*1024 + k
    int w   = gid >> 16;
    int rem = gid & 65535;
    int h   = rem >> 10;
    int k   = rem & 1023;
    const float* Wp = (w == 0) ? Wq : ((w == 1) ? Wk : Wv);
    wt[gid] = (_Float16)Wp[(size_t)k * H_ + h];
}

// ---------------------------------------------------------------------------
// Kernel 1: q/k/v projection.  grid = (B*L/16)/4 blocks of 128 threads.
// Each wave computes a 16-row tile of q,k,v (each 16x64) over K=D=1024.
// W tiles stream into LDS via async DMA, double buffered; x converts in-reg.
// v is written pre-transposed as vT[b][h][l] for the attention kernel.
// ---------------------------------------------------------------------------
__global__ __launch_bounds__(128) void proj_qkv_kernel(
    const float* __restrict__ x, const _Float16* __restrict__ wt,
    _Float16* __restrict__ qo, _Float16* __restrict__ ko,
    _Float16* __restrict__ vt)
{
    __shared__ __align__(32) _Float16 smW[2][3][H_][32];  // double-buffered WT tiles
    __shared__ __align__(32) _Float16 smX[4][16][32];     // per-wave x chunk

    const int t    = threadIdx.x;
    const int wave = t >> 5;
    const int lane = t & 31;
    const int nl   = lane & 15;
    const int hi   = lane >> 4;
    const size_t r0 = ((size_t)blockIdx.x * 4 + wave) * 16;   // tile start in B*L

    auto stageW = [&](int chunk, int buf) {
        #pragma unroll
        for (int i = 0; i < 6; ++i) {
            int idx = i * 128 + t;      // 768 x 16B = 12 KB (3 weights x 64 h x 64B)
            int row = idx >> 2;         // w*64 + h
            int c   = idx & 3;
            async_cp16(&smW[buf][0][0][0] + (size_t)row * 32 + c * 8,
                       wt + (size_t)row * D_ + chunk * 32 + c * 8);
        }
    };

    v8f acc[3][4];
    #pragma unroll
    for (int w = 0; w < 3; ++w)
        #pragma unroll
        for (int tt = 0; tt < 4; ++tt) acc[w][tt] = zero8();

    stageW(0, 0);
    wait_async0();
    __syncthreads();

    for (int it = 0; it < D_ / 32; ++it) {
        const int cur = it & 1;
        if (it + 1 < D_ / 32) stageW(it + 1, cur ^ 1);      // overlap next W tile

        // ---- per-wave stage of x chunk [16 x 32] f32 -> f16 (fused convert)
        const int kk = it * 32;
        #pragma unroll
        for (int i = 0; i < 4; ++i) {
            int idx4 = i * 32 + lane;
            int row  = idx4 >> 3;
            int c4   = idx4 & 7;
            const float4 f = *(const float4*)(x + (r0 + row) * D_ + kk + c4 * 4);
            h4 hv = { (_Float16)f.x, (_Float16)f.y, (_Float16)f.z, (_Float16)f.w };
            *(h4*)&smX[wave][row][c4 * 4] = hv;
        }

        // A fragment: lane = row nl, K contiguous per half-wave {hi*16..hi*16+15}
        AF a;
        a.h[0] = *(const h8*)&smX[wave][nl][hi * 16];
        a.h[1] = *(const h8*)&smX[wave][nl][hi * 16 + 8];

        #pragma unroll
        for (int w = 0; w < 3; ++w)
            #pragma unroll
            for (int tt = 0; tt < 4; ++tt) {
                // B fragment: lane = column h = tt*16+nl, K contiguous per half-wave
                v16h b = *(const v16h*)&smW[cur][w][tt * 16 + nl][hi * 16];
                acc[w][tt] = wmma_f16(a.v, b, acc[w][tt]);
            }

        wait_async0();          // next W tile has landed
        __syncthreads();        // everyone done reading smW[cur]
    }

    // epilogue: C frag row = hi*8 + j, col = tt*16 + nl
    #pragma unroll
    for (int tt = 0; tt < 4; ++tt)
        #pragma unroll
        for (int j = 0; j < 8; ++j) {
            int col = tt * 16 + nl;
            size_t grow = r0 + hi * 8 + j;          // global row in B*L
            qo[grow * H_ + col] = (_Float16)acc[0][tt][j];
            ko[grow * H_ + col] = (_Float16)acc[1][tt][j];
            int bb   = (int)(grow >> 11);           // / L_
            int lrow = (int)(grow & (L_ - 1));
            vt[((size_t)bb * H_ + col) * L_ + lrow] = (_Float16)acc[2][tt][j];
        }
}

// ---------------------------------------------------------------------------
// Kernel 2: flash attention.  grid = B*(L/128) blocks of 256 threads.
// Each wave owns 16 query rows; keys/values stream via async DMA into
// double-buffered LDS tiles (k natural [key][h], v pre-transposed [h][key]).
// ---------------------------------------------------------------------------
__global__ __launch_bounds__(256) void flash_attn_kernel(
    const _Float16* __restrict__ qw, const _Float16* __restrict__ kw,
    const _Float16* __restrict__ vt, const int* __restrict__ pad_mask,
    const int* __restrict__ attn_mask, float* __restrict__ out)
{
    __shared__ __align__(32) _Float16 smK[2][32][H_];   // keys natural
    __shared__ __align__(32) _Float16 smV[2][H_][32];   // values transposed
    __shared__ __align__(32) _Float16 smP[8][16][32];   // per-wave P tile

    const int t    = threadIdx.x;
    const int wave = t >> 5;
    const int lane = t & 31;
    const int nl   = lane & 15;
    const int hi   = lane >> 4;
    const int b    = blockIdx.x >> 4;
    const int q0   = (blockIdx.x & 15) * 128 + wave * 16;

    auto stageKV = [&](int chunk, int buf) {
        const int k0 = chunk * 32;
        // keys: 32x64 f16 = 4KB contiguous; one 16B async per thread
        async_cp16(&smK[buf][0][0] + t * 8,
                   kw + ((size_t)(b * L_) + k0) * H_ + t * 8);
        // values: 64 rows of 32 keys (pre-transposed in global)
        int h = t >> 2, c4 = t & 3;
        async_cp16(&smV[buf][h][c4 * 8],
                   vt + ((size_t)b * H_ + h) * L_ + k0 + c4 * 8);
    };

    // preload q A-fragments (16 rows x 64 -> two K=32 fragments)
    AF aq[2];
    {
        const _Float16* qp = qw + ((size_t)(b * L_) + q0 + nl) * H_;
        #pragma unroll
        for (int hh = 0; hh < 2; ++hh) {
            aq[hh].h[0] = *(const h8*)(qp + hh * 32 + hi * 16);
            aq[hh].h[1] = *(const h8*)(qp + hh * 32 + hi * 16 + 8);
        }
    }

    float m[8], ssum[8];
    v8f o[4];
    #pragma unroll
    for (int j = 0; j < 8; ++j) { m[j] = -3.0e38f; ssum[j] = 0.0f; }
    #pragma unroll
    for (int f = 0; f < 4; ++f) o[f] = zero8();

    const float scale = 0.03125f;   // D^-0.5 = 1/32

    stageKV(0, 0);
    wait_async0();
    __syncthreads();

    for (int c = 0; c < L_ / 32; ++c) {
        const int cur = c & 1;
        const int k0  = c * 32;
        if (c + 1 < L_ / 32) {
            stageKV(c + 1, cur ^ 1);                       // overlap next k/v tiles
            __builtin_prefetch(attn_mask + (size_t)(q0 + hi * 8) * L_ + k0 + 32, 0, 0);
        }

        // ---- S = q . k^T for two 16-key tiles
        float p[2][8];
        float rowm[8];
        #pragma unroll
        for (int j = 0; j < 8; ++j) rowm[j] = -3.0e38f;

        #pragma unroll
        for (int tt = 0; tt < 2; ++tt) {
            v8f s = zero8();
            #pragma unroll
            for (int hh = 0; hh < 2; ++hh) {
                v16h bk = *(const v16h*)&smK[cur][tt * 16 + nl][hh * 32 + hi * 16];
                s = wmma_f16(aq[hh].v, bk, s);
            }
            int kcol = k0 + tt * 16 + nl;
            int pv   = pad_mask[b * L_ + kcol];
            #pragma unroll
            for (int j = 0; j < 8; ++j) {
                int qrow = q0 + hi * 8 + j;
                float sv = s[j] * scale;
                int am = attn_mask[(size_t)qrow * L_ + kcol];
                if (am == 0 || pv == 0) sv = -3.0e38f;
                p[tt][j] = sv;
                rowm[j] = fmaxf(rowm[j], sv);
            }
        }

        // ---- online softmax update (rows live across 16-lane halves)
        #pragma unroll
        for (int j = 0; j < 8; ++j) {
            float v = rowm[j];
            #pragma unroll
            for (int msk = 1; msk < 16; msk <<= 1) v = fmaxf(v, __shfl_xor(v, msk, 32));
            float nm = fmaxf(m[j], v);
            float alpha = __expf(m[j] - nm);
            float r = 0.0f;
            #pragma unroll
            for (int tt = 0; tt < 2; ++tt) {
                float e = (p[tt][j] < -1.0e37f) ? 0.0f : __expf(p[tt][j] - nm);
                p[tt][j] = e;
                r += e;
            }
            #pragma unroll
            for (int msk = 1; msk < 16; msk <<= 1) r += __shfl_xor(r, msk, 32);
            ssum[j] = ssum[j] * alpha + r;
            m[j] = nm;
            #pragma unroll
            for (int f = 0; f < 4; ++f) o[f][j] *= alpha;
        }

        // ---- P through LDS (C-layout -> A-layout), then O += P . V
        #pragma unroll
        for (int tt = 0; tt < 2; ++tt)
            #pragma unroll
            for (int j = 0; j < 8; ++j)
                smP[wave][hi * 8 + j][tt * 16 + nl] = (_Float16)p[tt][j];

        AF ap;
        ap.h[0] = *(const h8*)&smP[wave][nl][hi * 16];
        ap.h[1] = *(const h8*)&smP[wave][nl][hi * 16 + 8];

        #pragma unroll
        for (int tv = 0; tv < 4; ++tv) {
            v16h bv = *(const v16h*)&smV[cur][tv * 16 + nl][hi * 16];
            o[tv] = wmma_f16(ap.v, bv, o[tv]);
        }

        wait_async0();          // next k/v tiles have landed
        __syncthreads();        // everyone done with smK/smV[cur]
    }

    // ---- normalize and store f32 output [B,L,H]
    #pragma unroll
    for (int j = 0; j < 8; ++j) {
        float inv = (ssum[j] > 0.0f) ? (1.0f / ssum[j]) : 0.0f;
        int row = q0 + hi * 8 + j;
        #pragma unroll
        for (int tv = 0; tv < 4; ++tv)
            out[((size_t)(b * L_) + row) * H_ + tv * 16 + nl] = o[tv][j] * inv;
    }
}

// ---------------------------------------------------------------------------
extern "C" void kernel_launch(void* const* d_in, const int* in_sizes, int n_in,
                              void* d_out, int out_size, void* d_ws, size_t ws_size,
                              hipStream_t stream) {
    const float* x  = (const float*)d_in[0];
    const float* Wq = (const float*)d_in[1];
    const float* Wk = (const float*)d_in[2];
    const float* Wv = (const float*)d_in[3];
    // d_in[4] = image_len (unused by the reference computation)
    const int* pad_mask  = (const int*)d_in[5];
    const int* attn_mask = (const int*)d_in[6];
    float* out = (float*)d_out;

    const size_t n = (size_t)B_ * L_ * H_;          // 1,048,576
    _Float16* qws = (_Float16*)d_ws;                // 2 MB
    _Float16* kws = qws + n;                        // 2 MB
    _Float16* vts = kws + n;                        // 2 MB (transposed v)
    _Float16* wts = vts + n;                        // 384 KB (transposed f16 W)

    convert_w_kernel<<<dim3(3 * H_ * D_ / 256), dim3(256), 0, stream>>>(
        Wq, Wk, Wv, wts);
    proj_qkv_kernel<<<dim3((B_ * L_ / 16) / 4), dim3(128), 0, stream>>>(
        x, wts, qws, kws, vts);
    flash_attn_kernel<<<dim3(B_ * (L_ / 128)), dim3(256), 0, stream>>>(
        qws, kws, vts, pad_mask, attn_mask, out);
}
